// RNNEncoder_50208167690375
// MI455X (gfx1250) — compile-verified, matrix operand order checked by reading, not run on hardware
//
#include <hip/hip_runtime.h>
#include <math.h>

// ---------------------------------------------------------------------------
// RNN encoder for MI455X (gfx1250), wave32, bf16 WMMA with f32 accumulation.
//   B=64, S=512, V=32000, E=1024, H=1024
// Phase 0: convert Wi, Wh f32 -> bf16 in workspace.
// Phase 1: xi[s*B+b, h] = emb[x[b,s],:] . Wi[h,:] + bi[h]   (big GEMM, WMMA,
//          4 n-tiles per wave for 4x A-fragment reuse)
// Phase 2: 512 sequential step kernels: h = tanh(xi_t + h . Wh^T + bh),
//          h state kept in bf16 (identical numerics: A is bf16-rounded before
//          the WMMA regardless), final step also emits f32 to d_out.
// ---------------------------------------------------------------------------

typedef __attribute__((ext_vector_type(16))) __bf16 v16bf;
typedef __attribute__((ext_vector_type(8)))  __bf16 v8bf;
typedef __attribute__((ext_vector_type(8)))  float  v8f;
typedef __attribute__((ext_vector_type(4)))  float  v4f;

static constexpr int BB = 64;     // batch
static constexpr int SS = 512;    // seq len
static constexpr int EE = 1024;   // embed dim
static constexpr int HH = 1024;   // hidden dim

// ---- fragment loaders -----------------------------------------------------
// 16-bit A/B fragment layout (ISA 7.12.2): lane&15 selects M (A) or N (B);
// element i maps to K = ((i>>3)<<4) + (half<<3) + (i&7), half = lane>>4.
// Each half-fragment is 8 consecutive K values -> one 16-byte load.

__device__ __forceinline__ v16bf load_frag_bf16(const __bf16* __restrict__ row, int kA) {
    v8bf lo = *(const v8bf*)(row + kA);        // K = kA .. kA+7
    v8bf hi = *(const v8bf*)(row + kA + 16);   // K = kA+16 .. kA+23
    return __builtin_shufflevector(lo, hi, 0,1,2,3,4,5,6,7,8,9,10,11,12,13,14,15);
}

__device__ __forceinline__ v16bf load_frag_f32(const float* __restrict__ row, int kA) {
    v4f a0 = *(const v4f*)(row + kA);
    v4f a1 = *(const v4f*)(row + kA + 4);
    v4f a2 = *(const v4f*)(row + kA + 16);
    v4f a3 = *(const v4f*)(row + kA + 20);
    v16bf r;
#pragma unroll
    for (int i = 0; i < 4; ++i) {
        r[i]      = (__bf16)a0[i];
        r[4 + i]  = (__bf16)a1[i];
        r[8 + i]  = (__bf16)a2[i];
        r[12 + i] = (__bf16)a3[i];
    }
    return r;
}

// ---- helper kernels -------------------------------------------------------

__global__ void cvt_f32_to_bf16(const float* __restrict__ src,
                                __bf16* __restrict__ dst, int n) {
    int i = blockIdx.x * blockDim.x + threadIdx.x;
    int stride = gridDim.x * blockDim.x;
    for (; i < n; i += stride) dst[i] = (__bf16)src[i];
}

__global__ void fill_zero_bf16(__bf16* __restrict__ p, int n) {
    int i = blockIdx.x * blockDim.x + threadIdx.x;
    if (i < n) p[i] = (__bf16)0.0f;
}

// ---- Phase 1: input projection GEMM ---------------------------------------
// grid = (H/64, S*B/64), block = 128 (4 waves).
// Block covers 64 rows x 64 cols. Each wave: one 16-row tile, FOUR 16-col
// WMMA tiles (A fragment reused 4x), K-loop over E in steps of 32.

__global__ void __launch_bounds__(128)
xi_gemm(const int*    __restrict__ x,     // [B,S] token ids
        const float*  __restrict__ emb,   // [V,E]
        const __bf16* __restrict__ Wi,    // [H,E] bf16
        const float*  __restrict__ bi,    // [H]
        float*        __restrict__ xi)    // [S*B,H]
{
    const int lane = threadIdx.x & 31;
    const int wave = threadIdx.x >> 5;
    const int half = lane >> 4;
    const int l15  = lane & 15;

    const int n0 = blockIdx.x * 64;              // column base (4 tiles)
    const int m0 = blockIdx.y * 64 + wave * 16;  // row base for this wave

    // A row for this lane: m = s*B + b  ->  b = m&63, s = m>>6
    const int m   = m0 + l15;
    const int b   = m & (BB - 1);
    const int s   = m >> 6;
    const int tok = x[b * SS + s];
    const float* arow = emb + (size_t)tok * EE;

    const __bf16* brow[4];
#pragma unroll
    for (int i = 0; i < 4; ++i)
        brow[i] = Wi + (size_t)(n0 + 16 * i + l15) * EE;

    v8f acc[4] = { {}, {}, {}, {} };

    for (int k0 = 0; k0 < EE; k0 += 32) {
        const int kA = k0 + half * 8;
        v16bf A = load_frag_f32(arow, kA);
#pragma unroll
        for (int i = 0; i < 4; ++i) {
            v16bf Bf = load_frag_bf16(brow[i], kA);
            acc[i] = __builtin_amdgcn_wmma_f32_16x16x32_bf16(
                false, A, false, Bf, (short)0, acc[i], false, false);
        }
    }

    // C/D layout: lane&15 = N; vgpr j = row (half*8 + j)
    const int rbase = m0 + half * 8;
#pragma unroll
    for (int i = 0; i < 4; ++i) {
        const int   c  = n0 + 16 * i + l15;
        const float bv = bi[c];
#pragma unroll
        for (int j = 0; j < 8; ++j)
            xi[(size_t)(rbase + j) * HH + c] = acc[i][j] + bv;
    }
}

// ---- Phase 2: one recurrent step ------------------------------------------
// grid = H/32 = 32 blocks, block = 128 (4 waves, 2 tiles each).
// h state is bf16 (A operand would be bf16-rounded anyway); f32 math for
// the xi_t add + tanh. Last step also writes f32 d_out.

__global__ void __launch_bounds__(128)
rnn_step(const __bf16* __restrict__ h_in,   // [B,H] bf16
         const __bf16* __restrict__ Wh,     // [H,H] bf16
         const float*  __restrict__ xi_t,   // [B,H] f32 (this timestep)
         const float*  __restrict__ bh,     // [H]
         __bf16*       __restrict__ h_out,  // [B,H] bf16
         float*        __restrict__ out_f32)// [B,H] f32 or nullptr
{
    const int lane = threadIdx.x & 31;
    const int wave = threadIdx.x >> 5;
    const int half = lane >> 4;
    const int l15  = lane & 15;

    const int n0 = blockIdx.x * 32;   // column base (2 tiles)
    const int m0 = wave * 16;         // row tile (B = 64 = 4 waves x 16)

    const __bf16* arow  = h_in + (size_t)(m0 + l15) * HH;
    const __bf16* b0row = Wh + (size_t)(n0 + l15) * HH;
    const __bf16* b1row = Wh + (size_t)(n0 + 16 + l15) * HH;

    v8f acc0 = {};
    v8f acc1 = {};

#pragma unroll 2
    for (int k0 = 0; k0 < HH; k0 += 32) {
        const int kA = k0 + half * 8;
        v16bf A  = load_frag_bf16(arow, kA);
        v16bf B0 = load_frag_bf16(b0row, kA);
        v16bf B1 = load_frag_bf16(b1row, kA);
        acc0 = __builtin_amdgcn_wmma_f32_16x16x32_bf16(false, A, false, B0,
                                                       (short)0, acc0, false, false);
        acc1 = __builtin_amdgcn_wmma_f32_16x16x32_bf16(false, A, false, B1,
                                                       (short)0, acc1, false, false);
    }

    const int   rbase = m0 + half * 8;
    const int   c0    = n0 + l15;
    const int   c1    = n0 + 16 + l15;
    const float bh0   = bh[c0];
    const float bh1   = bh[c1];
#pragma unroll
    for (int j = 0; j < 8; ++j) {
        const size_t r  = (size_t)(rbase + j) * HH;
        const float  t0 = tanhf(acc0[j] + xi_t[r + c0] + bh0);
        const float  t1 = tanhf(acc1[j] + xi_t[r + c1] + bh1);
        h_out[r + c0] = (__bf16)t0;
        h_out[r + c1] = (__bf16)t1;
        if (out_f32) {
            out_f32[r + c0] = t0;
            out_f32[r + c1] = t1;
        }
    }
}

// ---------------------------------------------------------------------------

extern "C" void kernel_launch(void* const* d_in, const int* in_sizes, int n_in,
                              void* d_out, int out_size, void* d_ws, size_t ws_size,
                              hipStream_t stream) {
    (void)in_sizes; (void)n_in; (void)out_size; (void)ws_size;

    const int*   x   = (const int*)  d_in[0];   // [B,S]
    const float* emb = (const float*)d_in[1];   // [V,E]
    const float* Wi  = (const float*)d_in[2];   // [H,E]
    const float* bi  = (const float*)d_in[3];   // [H]
    const float* Wh  = (const float*)d_in[4];   // [H,H]
    const float* bh  = (const float*)d_in[5];   // [H]
    float*       out = (float*)d_out;           // [B,H]

    // Workspace layout (~138.5 MB):
    //   Wi_bf [H*E] bf16 (2 MB) | Wh_bf [H*H] bf16 (2 MB)
    //   xi [S*B*H] f32 (134 MB) | h ping-pong 2x[B*H] bf16 (256 KB)
    char*   ws    = (char*)d_ws;
    __bf16* Wi_bf = (__bf16*)ws;
    __bf16* Wh_bf = (__bf16*)(ws + (size_t)HH * EE * 2);
    float*  xi    = (float*) (ws + (size_t)HH * EE * 2 * 2);
    __bf16* h0    = (__bf16*)((char*)xi + (size_t)SS * BB * HH * 4);
    __bf16* h1    = h0 + (size_t)BB * HH;

    // Phase 0: weight conversion + h0 = 0
    cvt_f32_to_bf16<<<1024, 256, 0, stream>>>(Wi, Wi_bf, HH * EE);
    cvt_f32_to_bf16<<<1024, 256, 0, stream>>>(Wh, Wh_bf, HH * HH);
    fill_zero_bf16<<<(BB * HH) / 256, 256, 0, stream>>>(h0, BB * HH);

    // Phase 1: input projection for all timesteps
    dim3 g1(HH / 64, (SS * BB) / 64);
    xi_gemm<<<g1, 128, 0, stream>>>(x, emb, Wi_bf, bi, xi);

    // Phase 2: sequential recurrence; last step also writes f32 d_out
    __bf16* bufs[2] = { h0, h1 };
    for (int t = 0; t < SS; ++t) {
        __bf16* hin  = bufs[t & 1];
        __bf16* hout = bufs[(t + 1) & 1];
        float*  of   = (t == SS - 1) ? out : nullptr;
        rnn_step<<<dim3(HH / 32), 128, 0, stream>>>(
            hin, Wh_bf, xi + (size_t)t * BB * HH, bh, hout, of);
    }
}
